// GCNN_desc_pool_25872882991626
// MI455X (gfx1250) — matrix operand order repaired and checked by reading, not compiled, not run on hardware
//
#include <hip/hip_runtime.h>

typedef __attribute__((ext_vector_type(16))) _Float16 v16h;
typedef __attribute__((ext_vector_type(8)))  float    v8f;

#define F_PRO 80
#define OUTC  128
#define BATCH 64
#define LSEQ  2048
#define COMB  (6 * OUTC)
#define SPLIT 4
#define NEG_SLOPE 0.01f

union HFrag { v16h v; _Float16 h[16]; };

__device__ __forceinline__ float lrelu_f(float x) { return x > 0.0f ? x : NEG_SLOPE * x; }

// Load the two guarded float4 runs of one 16x32 f16 A-fragment K-chunk.
// s0 run (K s0..s0+7) is always in range (max start 72); s1 run is zero-padded past D=80.
__device__ __forceinline__ void load_a_chunk(HFrag& a, const float* __restrict__ arow,
                                             int kc, bool hi) {
  int s0 = kc * 32 + (hi ? 8 : 0);
  float4 p0 = *(const float4*)(arow + s0);
  float4 p1 = *(const float4*)(arow + s0 + 4);
  a.h[0] = (_Float16)p0.x; a.h[1] = (_Float16)p0.y;
  a.h[2] = (_Float16)p0.z; a.h[3] = (_Float16)p0.w;
  a.h[4] = (_Float16)p1.x; a.h[5] = (_Float16)p1.y;
  a.h[6] = (_Float16)p1.z; a.h[7] = (_Float16)p1.w;
  int s1 = s0 + 16;
  if (s1 < F_PRO) {
    float4 q0 = *(const float4*)(arow + s1);
    float4 q1 = *(const float4*)(arow + s1 + 4);
    a.h[8]  = (_Float16)q0.x; a.h[9]  = (_Float16)q0.y;
    a.h[10] = (_Float16)q0.z; a.h[11] = (_Float16)q0.w;
    a.h[12] = (_Float16)q1.x; a.h[13] = (_Float16)q1.y;
    a.h[14] = (_Float16)q1.z; a.h[15] = (_Float16)q1.w;
  } else {
    #pragma unroll
    for (int j = 8; j < 16; ++j) a.h[j] = (_Float16)0.0f;
  }
}

// ---------------------------------------------------------------- utilities
__global__ void fill_kernel(float* __restrict__ p, float v, long long n) {
  long long i = (long long)blockIdx.x * blockDim.x + threadIdx.x;
  if (i < n) p[i] = v;
}

// ---------------------------------------------------------------- GCN: degree / dinv
__global__ void deg_kernel(const int* __restrict__ dst, float* __restrict__ deg, int E) {
  int e = blockIdx.x * blockDim.x + threadIdx.x;
  if (e < E) atomicAdd(&deg[dst[e]], 1.0f);
}

__global__ void dinv_kernel(float* __restrict__ deg, int N) {
  int i = blockIdx.x * blockDim.x + threadIdx.x;
  if (i < N) deg[i] = rsqrtf(fmaxf(deg[i], 1.0f));
}

// ---------------------------------------------------------------- XW = X @ Wg  (WMMA f16, f32 acc)
// Wg fragments staged in LDS; each wave owns one M-tile, loops all 5 N-tiles,
// so X is streamed exactly once. K padded 80 -> 96 (3 chunks of 32).
__global__ __launch_bounds__(256) void gemm_xw_wmma(
    const float* __restrict__ X, const float* __restrict__ Wg,
    float* __restrict__ XW, int M) {
  __shared__ __align__(16) _Float16 bsh[5][3][32][16];   // 7.5 KB

  int tid = threadIdx.x;
  int wave = tid >> 5, lane = tid & 31, n = lane & 15;
  bool hi = lane >= 16;

  // stage B fragments: frag(nt,kc), lane ln, slot j -> K = kc*32 + (ln>=16?8:0) + (j&7) + (j>=8?16:0)
  for (int idx = tid; idx < 5 * 3 * 32 * 16; idx += 256) {
    int j  = idx & 15;
    int ln = (idx >> 4) & 31;
    int kc = (idx >> 9) % 3;
    int nt = idx / 1536;
    int k   = kc * 32 + ((ln >= 16) ? 8 : 0) + (j & 7) + ((j & 8) ? 16 : 0);
    int col = nt * 16 + (ln & 15);
    bsh[nt][kc][ln][j] = (k < F_PRO) ? (_Float16)Wg[(size_t)k * F_PRO + col]
                                     : (_Float16)0.0f;
  }
  __syncthreads();

  int mt = blockIdx.x * 8 + wave;
  if (mt * 16 >= M) return;

  const float* arow = X + (size_t)(mt * 16 + n) * F_PRO;
  HFrag af[3];
  #pragma unroll
  for (int kc = 0; kc < 3; ++kc) load_a_chunk(af[kc], arow, kc, hi);

  int m0 = mt * 16 + (hi ? 8 : 0);
  #pragma unroll
  for (int nt = 0; nt < 5; ++nt) {
    v8f acc = {0.f, 0.f, 0.f, 0.f, 0.f, 0.f, 0.f, 0.f};
    #pragma unroll
    for (int kc = 0; kc < 3; ++kc) {
      HFrag bf;
      const uint4* bp = (const uint4*)&bsh[nt][kc][lane][0];
      *(uint4*)&bf.h[0] = bp[0];
      *(uint4*)&bf.h[8] = bp[1];
      acc = __builtin_amdgcn_wmma_f32_16x16x32_f16(false, af[kc].v, false, bf.v,
                                                   (short)0, acc, false, false);
    }
    #pragma unroll
    for (int r = 0; r < 8; ++r)
      XW[(size_t)(m0 + r) * F_PRO + nt * 16 + n] = acc[r];
  }
}

// ---------------------------------------------------------------- edge scatter
__global__ void scatter_kernel(const int* __restrict__ src, const int* __restrict__ dst,
                               const float* __restrict__ xw, const float* __restrict__ dinv,
                               float* __restrict__ hacc, int E) {
  long long tid = (long long)blockIdx.x * blockDim.x + threadIdx.x;
  int e = (int)(tid / 20);
  int c = (int)(tid % 20);
  if (e >= E) return;
  int s = src[e], d = dst[e];
  float coef = dinv[s] * dinv[d];
  float4 v = *(const float4*)(xw + (size_t)s * F_PRO + c * 4);
  float* o = hacc + (size_t)d * F_PRO + c * 4;
  atomicAdd(o + 0, v.x * coef);
  atomicAdd(o + 1, v.y * coef);
  atomicAdd(o + 2, v.z * coef);
  atomicAdd(o + 3, v.w * coef);
}

// ---------------------------------------------------------------- self-loop + bias + lrelu + pooling
__global__ void selfpool_kernel(const float* __restrict__ xw, const float* __restrict__ dinv,
                                const float* __restrict__ hacc, const float* __restrict__ bias,
                                const int* __restrict__ batch, float* __restrict__ pool, int N) {
  long long tid = (long long)blockIdx.x * blockDim.x + threadIdx.x;
  int i = (int)(tid / 20);
  int c = (int)(tid % 20);
  if (i >= N) return;
  float di = dinv[i];
  float coef = di * di;
  float4 xv = *(const float4*)(xw   + (size_t)i * F_PRO + c * 4);
  float4 hv = *(const float4*)(hacc + (size_t)i * F_PRO + c * 4);
  float4 bv = *(const float4*)(bias + c * 4);
  float r0 = lrelu_f(hv.x + xv.x * coef + bv.x);
  float r1 = lrelu_f(hv.y + xv.y * coef + bv.y);
  float r2 = lrelu_f(hv.z + xv.z * coef + bv.z);
  float r3 = lrelu_f(hv.w + xv.w * coef + bv.w);
  int b = batch[i];
  float* o = pool + (size_t)b * F_PRO + c * 4;
  atomicAdd(o + 0, r0);
  atomicAdd(o + 1, r1);
  atomicAdd(o + 2, r2);
  atomicAdd(o + 3, r3);
}

__global__ void count_kernel(const int* __restrict__ batch, float* __restrict__ cnt, int N) {
  int i = blockIdx.x * blockDim.x + threadIdx.x;
  if (i < N) atomicAdd(&cnt[batch[i]], 1.0f);
}

// ---------------------------------------------------------------- FC
__global__ void fc_kernel(const float* __restrict__ pool, const float* __restrict__ cnt,
                          const float* __restrict__ Wfc, const float* __restrict__ bfc,
                          float* __restrict__ comb, int off) {
  int tid = blockIdx.x * blockDim.x + threadIdx.x;
  if (tid >= BATCH * OUTC) return;
  int b = tid / OUTC, o = tid % OUTC;
  float inv = 1.0f / fmaxf(cnt[b], 1.0f);
  float s = bfc[o];
  for (int d = 0; d < F_PRO; ++d)
    s += pool[b * F_PRO + d] * inv * Wfc[d * OUTC + o];
  comb[(size_t)b * COMB + off + o] = lrelu_f(s);
}

// ---------------------------------------------------------------- descriptor branch
// All 8x3 B-fragments of Wc staged in LDS (24 KB); each wave keeps its A-fragments
// in registers across the 8 N-tiles, so mas is streamed exactly once.
// Grid: (BATCH, SPLIT); each block covers L/SPLIT rows; partial maxima -> part slab.
__global__ __launch_bounds__(256) void desc_kernel(
    const float* __restrict__ mas, const float* __restrict__ Wc,
    const float* __restrict__ bc, const int* __restrict__ lens,
    float* __restrict__ part, int qoff) {
  __shared__ __align__(16) _Float16 bsh[8][3][32][16];   // 24 KB
  __shared__ float red[8][8][16];                        // waves x nt x col

  int b   = blockIdx.x;
  int sp  = blockIdx.y;
  int tid = threadIdx.x;
  int wave = tid >> 5, lane = tid & 31, n = lane & 15;
  bool hi = lane >= 16;

  // stage B fragments for all 8 N-tiles x 3 K-chunks
  for (int idx = tid; idx < 8 * 3 * 32 * 16; idx += 256) {
    int j  = idx & 15;
    int ln = (idx >> 4) & 31;
    int kc = (idx >> 9) % 3;
    int nt = idx / 1536;
    int k   = kc * 32 + ((ln >= 16) ? 8 : 0) + (j & 7) + ((j & 8) ? 16 : 0);
    int row = nt * 16 + (ln & 15);
    bsh[nt][kc][ln][j] = (k < F_PRO) ? (_Float16)Wc[(size_t)row * F_PRO + k]
                                     : (_Float16)0.0f;
  }

  int len = lens[b];
  float bnv[8];
  #pragma unroll
  for (int nt = 0; nt < 8; ++nt) bnv[nt] = bc[nt * 16 + n];

  __syncthreads();

  const int MT_PER_SPLIT = (LSEQ / 16) / SPLIT;   // 32
  int mtBase = sp * MT_PER_SPLIT;
  const float* masb = mas + (size_t)b * LSEQ * F_PRO;

  float mx[8];
  #pragma unroll
  for (int nt = 0; nt < 8; ++nt) mx[nt] = -INFINITY;

  for (int mti = wave; mti < MT_PER_SPLIT; mti += 8) {
    int l0 = (mtBase + mti) * 16;
    const float* arow = masb + (size_t)(l0 + n) * F_PRO;
    if (mti + 8 < MT_PER_SPLIT)
      __builtin_prefetch(masb + (size_t)(l0 + 128 + n) * F_PRO, 0, 1);

    HFrag af[3];
    #pragma unroll
    for (int kc = 0; kc < 3; ++kc) load_a_chunk(af[kc], arow, kc, hi);

    #pragma unroll
    for (int nt = 0; nt < 8; ++nt) {
      v8f acc = {0.f, 0.f, 0.f, 0.f, 0.f, 0.f, 0.f, 0.f};
      #pragma unroll
      for (int kc = 0; kc < 3; ++kc) {
        HFrag bf;
        const uint4* bp = (const uint4*)&bsh[nt][kc][lane][0];
        *(uint4*)&bf.h[0] = bp[0];
        *(uint4*)&bf.h[8] = bp[1];
        acc = __builtin_amdgcn_wmma_f32_16x16x32_f16(false, af[kc].v, false, bf.v,
                                                     (short)0, acc, false, false);
      }
      #pragma unroll
      for (int r = 0; r < 8; ++r) {
        int l = l0 + r + (hi ? 8 : 0);
        float v = acc[r] + bnv[nt];
        v = v > 0.0f ? v : NEG_SLOPE * v;
        v = (l < len) ? v : -INFINITY;
        mx[nt] = fmaxf(mx[nt], v);
      }
    }
  }

  // fold half-waves, then cross-wave reduce in LDS
  #pragma unroll
  for (int nt = 0; nt < 8; ++nt) {
    float m = fmaxf(mx[nt], __shfl_xor(mx[nt], 16, 32));
    if (lane < 16) red[wave][nt][n] = m;
  }
  __syncthreads();
  if (tid < 128) {
    int nt = tid >> 4, c = tid & 15;
    float m = red[0][nt][c];
    #pragma unroll
    for (int w = 1; w < 8; ++w) m = fmaxf(m, red[w][nt][c]);
    part[((size_t)sp * BATCH + b) * (4 * OUTC) + qoff + nt * 16 + c] = m;
  }
}

// max over the SPLIT partial slabs -> combined slab
__global__ void desc_reduce_kernel(const float* __restrict__ part, float* __restrict__ comb) {
  int tid = blockIdx.x * blockDim.x + threadIdx.x;
  if (tid >= BATCH * 4 * OUTC) return;
  int b = tid / (4 * OUTC), c = tid % (4 * OUTC);
  float m = part[tid];   // split 0
  #pragma unroll
  for (int s = 1; s < SPLIT; ++s)
    m = fmaxf(m, part[((size_t)s * BATCH + b) * (4 * OUTC) + c]);
  comb[(size_t)b * COMB + 2 * OUTC + c] = m;
}

// ---------------------------------------------------------------- final dot
__global__ void final_kernel(const float* __restrict__ comb, const float* __restrict__ Wf,
                             const float* __restrict__ bf, float* __restrict__ out) {
  int b = threadIdx.x;
  if (b >= BATCH) return;
  float s = bf[0];
  for (int j = 0; j < COMB; ++j) s += comb[(size_t)b * COMB + j] * Wf[j];
  out[b] = s;
}

// ================================================================ host launcher
extern "C" void kernel_launch(void* const* d_in, const int* in_sizes, int n_in,
                              void* d_out, int out_size, void* d_ws, size_t ws_size,
                              hipStream_t stream) {
  const float* pro_x[2]   = {(const float*)d_in[0], (const float*)d_in[3]};
  const int*   pro_ei[2]  = {(const int*)d_in[1],   (const int*)d_in[4]};
  const int*   pro_bat[2] = {(const int*)d_in[2],   (const int*)d_in[5]};
  const float* mas[4]  = {(const float*)d_in[6], (const float*)d_in[7],
                          (const float*)d_in[8], (const float*)d_in[9]};
  const int*   lens[4] = {(const int*)d_in[10], (const int*)d_in[11],
                          (const int*)d_in[12], (const int*)d_in[13]};
  const float* Wg[2]  = {(const float*)d_in[14], (const float*)d_in[18]};
  const float* bg[2]  = {(const float*)d_in[15], (const float*)d_in[19]};
  const float* Wfc[2] = {(const float*)d_in[16], (const float*)d_in[20]};
  const float* bfc[2] = {(const float*)d_in[17], (const float*)d_in[21]};
  const float* Wc[4]  = {(const float*)d_in[22], (const float*)d_in[24],
                         (const float*)d_in[26], (const float*)d_in[28]};
  const float* bcv[4] = {(const float*)d_in[23], (const float*)d_in[25],
                         (const float*)d_in[27], (const float*)d_in[29]};
  const float* Wf = (const float*)d_in[30];
  const float* bf = (const float*)d_in[31];

  const int N = in_sizes[0] / F_PRO;    // 32000
  const int E = in_sizes[1] / 2;        // 512000

  // workspace slabs (floats)
  float* ws = (float*)d_ws;
  size_t off = 0;
  float* deg  = ws + off; off += (size_t)N;                       // degree -> dinv
  float* xw   = ws + off; off += (size_t)N * F_PRO;               // X @ Wg
  float* hacc = ws + off; off += (size_t)N * F_PRO;               // neighbor accum
  float* pool = ws + off; off += (size_t)BATCH * F_PRO;           // segment sums
  float* cnt  = ws + off; off += (size_t)BATCH;                   // segment counts
  float* comb = ws + off; off += (size_t)BATCH * COMB;            // [64, 768]
  float* part = ws + off; off += (size_t)SPLIT * BATCH * 4 * OUTC; // desc partial maxima

  const int TB = 256;
  for (int p = 0; p < 2; ++p) {
    long long nh = (long long)N * F_PRO;
    fill_kernel<<<(unsigned)((N + TB - 1) / TB), TB, 0, stream>>>(deg, 1.0f, N);
    fill_kernel<<<(unsigned)((nh + TB - 1) / TB), TB, 0, stream>>>(hacc, 0.0f, nh);
    fill_kernel<<<(BATCH * F_PRO + TB - 1) / TB, TB, 0, stream>>>(pool, 0.0f, BATCH * F_PRO);
    fill_kernel<<<1, TB, 0, stream>>>(cnt, 0.0f, BATCH);

    deg_kernel<<<(E + TB - 1) / TB, TB, 0, stream>>>(pro_ei[p] + E, deg, E);
    dinv_kernel<<<(N + TB - 1) / TB, TB, 0, stream>>>(deg, N);

    int mtiles = N / 16;                              // 2000
    gemm_xw_wmma<<<(mtiles + 7) / 8, 256, 0, stream>>>(pro_x[p], Wg[p], xw, N);

    long long st = (long long)E * 20;
    scatter_kernel<<<(unsigned)((st + TB - 1) / TB), TB, 0, stream>>>(
        pro_ei[p], pro_ei[p] + E, xw, deg, hacc, E);

    long long sp = (long long)N * 20;
    selfpool_kernel<<<(unsigned)((sp + TB - 1) / TB), TB, 0, stream>>>(
        xw, deg, hacc, bg[p], pro_bat[p], pool, N);
    count_kernel<<<(N + TB - 1) / TB, TB, 0, stream>>>(pro_bat[p], cnt, N);

    fc_kernel<<<(BATCH * OUTC + TB - 1) / TB, TB, 0, stream>>>(
        pool, cnt, Wfc[p], bfc[p], comb, p * OUTC);
  }

  dim3 dgrid(BATCH, SPLIT);
  for (int q = 0; q < 4; ++q)
    desc_kernel<<<dgrid, 256, 0, stream>>>(mas[q], Wc[q], bcv[q], lens[q],
                                           part, q * OUTC);
  desc_reduce_kernel<<<(BATCH * 4 * OUTC + TB - 1) / TB, TB, 0, stream>>>(part, comb);

  final_kernel<<<1, 64, 0, stream>>>(comb, Wf, bf, (float*)d_out);
}
